// HopfieldLayer_27075473834156
// MI455X (gfx1250) — matrix operand order repaired and checked by reading, not compiled, not run on hardware
//
#include <hip/hip_runtime.h>
#include <hip/hip_bf16.h>

// ---------------------------------------------------------------------------
// CDNA5 (gfx1250) bf16-WMMA Hopfield / MHA layer.
//   B=4, L=S=1024, D_MODEL=1024, H=16, Dh=64.
// Matrix math: v_wmma_f32_16x16x32_bf16 (f32 accumulate).
// All LDS fragment traffic is contiguous 16B (ds_load_b128 / ds_store_b128);
// K/V tile staging uses async global->LDS (double buffered) when available.
// ---------------------------------------------------------------------------

typedef __bf16 bf16;
typedef bf16  v16bf __attribute__((ext_vector_type(16)));
typedef float v8f   __attribute__((ext_vector_type(8)));
typedef int   v4i   __attribute__((ext_vector_type(4)));

#define BATCH   4
#define SEQ     1024
#define DMODEL  1024
#define NHEADS  16
#define DHEAD   64
#define ROWS    (BATCH * SEQ)       // 4096

// ---------------- CDNA5 feature probes --------------------------------------
#if __has_builtin(__builtin_amdgcn_global_load_async_to_lds_b128) && \
    __has_builtin(__builtin_amdgcn_s_wait_asynccnt)
#define ASYNC_LDS 1
#endif
#if __has_builtin(__builtin_amdgcn_mov_dpp8)
#define HAVE_DPP8 1
#endif
#if __has_builtin(__builtin_amdgcn_permlane16)
#define HAVE_PL16 1
#endif

typedef __attribute__((address_space(1))) v4i gv4i;   // global <4 x i32>
typedef __attribute__((address_space(3))) v4i lv4i;   // LDS    <4 x i32>

// 16-byte global -> LDS copy (async when the gfx1250 builtin exists).
__device__ __forceinline__ void cp16_g2s(bf16* dst, const bf16* src) {
#ifdef ASYNC_LDS
    __builtin_amdgcn_global_load_async_to_lds_b128((gv4i*)src, (lv4i*)dst, 0, 0);
#else
    *(uint4*)dst = *(const uint4*)src;
#endif
}
template <int N>
__device__ __forceinline__ void wait_cp() {
#ifdef ASYNC_LDS
    __builtin_amdgcn_s_wait_asynccnt(N);
#endif
}

// ---------------- wave32 row reductions (16-lane halves) ---------------------
#define DPP8_SEL(a,b,c,d,e,f,g,h) \
    ((a)|((b)<<3)|((c)<<6)|((d)<<9)|((e)<<12)|((f)<<15)|((g)<<18)|((h)<<21))

__device__ __forceinline__ float xor8_lane(float x) {
#ifdef HAVE_PL16
    // xor-8 inside each 16-lane group: per-nibble selects {8..15,0..7}
    return __int_as_float(__builtin_amdgcn_permlane16(
        __float_as_int(x), __float_as_int(x), 0xFEDCBA98u, 0x76543210u,
        false, false));
#else
    return __shfl_xor(x, 8);
#endif
}
__device__ __forceinline__ float row_max16(float x) {
#ifdef HAVE_DPP8
    x = fmaxf(x, __int_as_float(__builtin_amdgcn_mov_dpp8(__float_as_int(x), DPP8_SEL(1,0,3,2,5,4,7,6))));
    x = fmaxf(x, __int_as_float(__builtin_amdgcn_mov_dpp8(__float_as_int(x), DPP8_SEL(2,3,0,1,6,7,4,5))));
    x = fmaxf(x, __int_as_float(__builtin_amdgcn_mov_dpp8(__float_as_int(x), DPP8_SEL(4,5,6,7,0,1,2,3))));
#else
    x = fmaxf(x, __shfl_xor(x, 1));
    x = fmaxf(x, __shfl_xor(x, 2));
    x = fmaxf(x, __shfl_xor(x, 4));
#endif
    return fmaxf(x, xor8_lane(x));
}
__device__ __forceinline__ float row_sum16(float x) {
#ifdef HAVE_DPP8
    x += __int_as_float(__builtin_amdgcn_mov_dpp8(__float_as_int(x), DPP8_SEL(1,0,3,2,5,4,7,6)));
    x += __int_as_float(__builtin_amdgcn_mov_dpp8(__float_as_int(x), DPP8_SEL(2,3,0,1,6,7,4,5)));
    x += __int_as_float(__builtin_amdgcn_mov_dpp8(__float_as_int(x), DPP8_SEL(4,5,6,7,0,1,2,3)));
#else
    x += __shfl_xor(x, 1);
    x += __shfl_xor(x, 2);
    x += __shfl_xor(x, 4);
#endif
    return x + xor8_lane(x);
}

// ---------------- WMMA helpers ----------------------------------------------
__device__ __forceinline__ v8f wmma_bf16(v16bf a, v16bf b, v8f c) {
    return __builtin_amdgcn_wmma_f32_16x16x32_bf16(false, a, false, b,
                                                   (short)0, c, false, false);
}

// A-frag (16x32 bf16, ISA 7.12.2): lane m=lane&15, kh=lane>>4.
// Element i holds k = 16*kh + i (matches B-frag order).
// p -> row m of a row-major tile  =>  two contiguous 16B loads.
__device__ __forceinline__ v16bf load_afrag16(const bf16* p, int kh) {
    union { v16bf v; uint4 q[2]; } r;
    r.q[0] = *(const uint4*)(p + 16 * kh);
    r.q[1] = *(const uint4*)(p + 16 * kh + 8);
    return r.v;
}
// B-frag (32x16 bf16): lane n=lane&15, element i holds k = 16*kh + i.
// p -> the lane's 16 contiguous k-values  =>  two contiguous 16B loads.
__device__ __forceinline__ v16bf load_bfrag16(const bf16* p) {
    union { v16bf v; uint4 q[2]; } r;
    r.q[0] = *(const uint4*)(p);
    r.q[1] = *(const uint4*)(p + 8);
    return r.v;
}

// ===========================================================================
// Tiled GEMM:  C[M,N] = A[M,K] * W[K,N] + bias
//   block = 256 thr (8 waves, 4x2), tile 128(M) x 64(N), K step 32.
//   A tile row-major bf16 in LDS; W tile stored TRANSPOSED (Wt[n][k]) so both
//   fragment loads are contiguous b128.  OUT_TRANS stores C^T (for V).
// ===========================================================================
#define ASTR 40   // k-stride of A tile rows   (80B, 16B aligned)
#define WSTR 40   // k-stride of Wt rows       (80B, 16B aligned)

template <bool IN_BF16, bool OUT_BF16, bool OUT_TRANS>
__global__ __launch_bounds__(256) void gemm_wmma(const void* __restrict__ Ap,
                                                 const float* __restrict__ W,
                                                 const float* __restrict__ bias,
                                                 void* __restrict__ Cp,
                                                 int M, int N, int K) {
    const int tid  = threadIdx.x;
    const int lane = tid & 31;
    const int w    = tid >> 5;      // 0..7
    const int wm   = w >> 1;        // 0..3 : 32-row strip
    const int wn   = w & 1;         // 0..1 : 32-col strip
    const int m    = lane & 15;
    const int kh   = lane >> 4;
    const int mBlk = blockIdx.x * 128;
    const int nBlk = blockIdx.y * 64;

    __shared__ bf16 As[128 * ASTR];
    __shared__ bf16 Wt[64 * WSTR];

    v8f acc[2][2] = {};

    const int arow = tid >> 1;          // 0..127
    const int ac0  = (tid & 1) * 16;    // 0,16
    const int wcol = tid & 63;          // n within tile
    const int wkc  = (tid >> 6) * 8;    // k chunk 0,8,16,24

    for (int kt = 0; kt < K; kt += 32) {
        // ---- stage A tile (row-major bf16) ----
        if (IN_BF16) {
            const bf16* src = (const bf16*)Ap + (size_t)(mBlk + arow) * K + kt + ac0;
            cp16_g2s(As + arow * ASTR + ac0,     src);
            cp16_g2s(As + arow * ASTR + ac0 + 8, src + 8);
        } else {
            const float* src = (const float*)Ap + (size_t)(mBlk + arow) * K + kt + ac0;
            bf16* dst = As + arow * ASTR + ac0;
#pragma unroll
            for (int i = 0; i < 16; ++i) dst[i] = (bf16)src[i];
        }
        // ---- stage W tile transposed: Wt[n][k] ----
        // Each lane reads a column of W (lane-consecutive n => coalesced),
        // writes one contiguous 16B LDS chunk.
        {
            union { uint4 q; bf16 e[8]; } pk;
#pragma unroll
            for (int i = 0; i < 8; ++i)
                pk.e[i] = (bf16)W[(size_t)(kt + wkc + i) * N + nBlk + wcol];
            *(uint4*)(Wt + wcol * WSTR + wkc) = pk.q;
        }
        if (kt + 32 < K) {
            __builtin_prefetch(W + (size_t)(kt + 32 + wkc) * N + nBlk + wcol, 0, 0);
            if (!IN_BF16)
                __builtin_prefetch((const float*)Ap + (size_t)(mBlk + arow) * K + kt + 32 + ac0, 0, 0);
        }
        wait_cp<0>();
        __syncthreads();

        const v16bf a0 = load_afrag16(As + (wm * 32 + m) * ASTR, kh);
        const v16bf a1 = load_afrag16(As + (wm * 32 + 16 + m) * ASTR, kh);
        const v16bf b0 = load_bfrag16(Wt + (wn * 32 + m) * WSTR + kh * 16);
        const v16bf b1 = load_bfrag16(Wt + (wn * 32 + 16 + m) * WSTR + kh * 16);
        acc[0][0] = wmma_bf16(a0, b0, acc[0][0]);
        acc[0][1] = wmma_bf16(a0, b1, acc[0][1]);
        acc[1][0] = wmma_bf16(a1, b0, acc[1][0]);
        acc[1][1] = wmma_bf16(a1, b1, acc[1][1]);
        __syncthreads();
    }

    // ---- epilogue : C/D layout row = r + 8*kh, col = lane&15 ----
#pragma unroll
    for (int rt = 0; rt < 2; ++rt) {
#pragma unroll
        for (int ct = 0; ct < 2; ++ct) {
            const int col  = nBlk + wn * 32 + ct * 16 + m;
            const int row0 = mBlk + wm * 32 + rt * 16 + 8 * kh;
            const float bb = bias[col];
            if (OUT_TRANS) {
                // col-major store: 8 consecutive rows per lane -> one b128
                union { uint4 q; bf16 e[8]; } pk;
#pragma unroll
                for (int r = 0; r < 8; ++r) pk.e[r] = (bf16)(acc[rt][ct][r] + bb);
                *(uint4*)((bf16*)Cp + (size_t)col * M + row0) = pk.q;
            } else {
#pragma unroll
                for (int r = 0; r < 8; ++r) {
                    const float v = acc[rt][ct][r] + bb;
                    const size_t idx = (size_t)(row0 + r) * N + col;
                    if (OUT_BF16) ((bf16*)Cp)[idx] = (bf16)v;
                    else          ((float*)Cp)[idx] = v;
                }
            }
        }
    }
}

// ===========================================================================
// Flash attention: one block per (b, h, 128 query rows); 8 waves x 16 q-rows.
// K tiles row-major [32 x 64]; V supplied TRANSPOSED in global (VT[h*64+d][b*S+s])
// so both K and V staging are straight 16B async copies and all fragment
// loads are contiguous b128.  Double-buffered tiles overlap copy with WMMA.
// ===========================================================================
#define KSTR 72   // Ks row stride (144B, 16B aligned)
#define VSTR 40   // Vt row stride (80B, 16B aligned)

__global__ __launch_bounds__(256) void hopfield_attn(const bf16* __restrict__ Q,
                                                     const bf16* __restrict__ Km,
                                                     const bf16* __restrict__ VT,
                                                     bf16* __restrict__ O) {
    const int tid  = threadIdx.x;
    const int lane = tid & 31;
    const int w    = tid >> 5;
    const int m    = lane & 15;
    const int kh   = lane >> 4;
    const int bh   = blockIdx.y;       // 0..63
    const int b    = bh >> 4;
    const int h    = bh & 15;
    const int q0   = blockIdx.x * 128 + w * 16;

    __shared__ bf16 Ks[2 * 32 * KSTR];     // double-buffered K tiles [s][e]
    __shared__ bf16 Vt[2 * 64 * VSTR];     // double-buffered V^T tiles [d][s]
    __shared__ bf16 Ps[8 * 16 * 32];       // per-wave P re-layout buffer
    bf16* Pw = Ps + w * 16 * 32;

    // Q fragments (resident): rows q0+m, cols h*64 + {0..31, 32..63}
    const size_t qoff = (size_t)(b * SEQ + q0 + m) * DMODEL + h * DHEAD;
    const v16bf qa0 = load_afrag16(Q + qoff, kh);
    const v16bf qa1 = load_afrag16(Q + qoff + 32, kh);

    v8f o0 = {}, o1 = {}, o2 = {}, o3 = {};
    float rmax[8], rsum[8];
#pragma unroll
    for (int r = 0; r < 8; ++r) { rmax[r] = -3.0e38f; rsum[r] = 0.0f; }

    // staging assignments
    const int krow = tid >> 3;          // 0..31 (s)
    const int kc   = (tid & 7) * 8;     // e chunk
    const int vd   = tid & 63;          // d
    const int vsc  = (tid >> 6) * 8;    // s chunk

    const bf16* kbase = Km + (size_t)b * SEQ * DMODEL + h * DHEAD;
    const bf16* vbase = VT + (size_t)(h * DHEAD) * ROWS + (size_t)b * SEQ;

    // issue tile j into buffer (j&1): 2 async ops per thread
    auto issue = [&](int j) {
        const int st = j * 32;
        bf16* ksb = Ks + (j & 1) * 32 * KSTR;
        bf16* vtb = Vt + (j & 1) * 64 * VSTR;
        cp16_g2s(ksb + krow * KSTR + kc, kbase + (size_t)(st + krow) * DMODEL + kc);
        cp16_g2s(vtb + vd * VSTR + vsc,  vbase + (size_t)vd * ROWS + st + vsc);
    };

    issue(0);
    const int NT = SEQ / 32;            // 32 tiles
    for (int j = 0; j < NT; ++j) {
        const bf16* ksb = Ks + (j & 1) * 32 * KSTR;
        const bf16* vtb = Vt + (j & 1) * 64 * VSTR;

        __syncthreads();                 // everyone done computing on buf j^1
        if (j + 1 < NT) { issue(j + 1); wait_cp<2>(); }
        else            { wait_cp<0>(); }
        __syncthreads();                 // tile j visible to all waves

        // ---- scores S[16x32] = Q * K^T : B(k=e,n=s)=K[s][e], contiguous ----
        v8f sc0 = {}, sc1 = {};
        {
            const v16bf kb00 = load_bfrag16(ksb + (m)      * KSTR + kh * 16);
            const v16bf kb01 = load_bfrag16(ksb + (m)      * KSTR + 32 + kh * 16);
            const v16bf kb10 = load_bfrag16(ksb + (16 + m) * KSTR + kh * 16);
            const v16bf kb11 = load_bfrag16(ksb + (16 + m) * KSTR + 32 + kh * 16);
            sc0 = wmma_bf16(qa0, kb00, sc0);
            sc0 = wmma_bf16(qa1, kb01, sc0);
            sc1 = wmma_bf16(qa0, kb10, sc1);
            sc1 = wmma_bf16(qa1, kb11, sc1);
        }

        // ---- online softmax over this 32-wide slab (DPP8/permlane) ----
#pragma unroll
        for (int r = 0; r < 8; ++r) {
            const float x0 = sc0[r] * 0.125f;        // 1/sqrt(64)
            const float x1 = sc1[r] * 0.125f;
            const float mx = row_max16(fmaxf(x0, x1));
            const float nm = fmaxf(rmax[r], mx);
            const float alpha = __expf(rmax[r] - nm);
            rmax[r] = nm;
            const float p0 = __expf(x0 - nm);
            const float p1 = __expf(x1 - nm);
            rsum[r] = rsum[r] * alpha + row_sum16(p0 + p1);
            o0[r] *= alpha; o1[r] *= alpha; o2[r] *= alpha; o3[r] *= alpha;
            // D-layout (row r+8kh, col m) -> row-major P for A-frag reload
            Pw[(r + 8 * kh) * 32 + m]      = (bf16)p0;
            Pw[(r + 8 * kh) * 32 + 16 + m] = (bf16)p1;
        }

        // ---- O += P[16x32] * V[32x64] ; V-frags contiguous from Vt[d][s] ----
        const v16bf pa  = load_afrag16(Pw + m * 32, kh);
        const v16bf vb0 = load_bfrag16(vtb + (m)      * VSTR + kh * 16);
        const v16bf vb1 = load_bfrag16(vtb + (16 + m) * VSTR + kh * 16);
        const v16bf vb2 = load_bfrag16(vtb + (32 + m) * VSTR + kh * 16);
        const v16bf vb3 = load_bfrag16(vtb + (48 + m) * VSTR + kh * 16);
        o0 = wmma_bf16(pa, vb0, o0);
        o1 = wmma_bf16(pa, vb1, o1);
        o2 = wmma_bf16(pa, vb2, o2);
        o3 = wmma_bf16(pa, vb3, o3);
    }

    // ---- normalize + store attention output (bf16, row-major) ----
#pragma unroll
    for (int r = 0; r < 8; ++r) {
        const float inv = 1.0f / rsum[r];
        const size_t base = (size_t)(b * SEQ + q0 + r + 8 * kh) * DMODEL + h * DHEAD;
        O[base + m]      = (bf16)(o0[r] * inv);
        O[base + 16 + m] = (bf16)(o1[r] * inv);
        O[base + 32 + m] = (bf16)(o2[r] * inv);
        O[base + 48 + m] = (bf16)(o3[r] * inv);
    }
}

// ===========================================================================
// Host-side launch
// ===========================================================================
extern "C" void kernel_launch(void* const* d_in, const int* in_sizes, int n_in,
                              void* d_out, int out_size, void* d_ws, size_t ws_size,
                              hipStream_t stream) {
    const float* queries = (const float*)d_in[0];
    const float* keys    = (const float*)d_in[1];
    const float* values  = (const float*)d_in[2];
    const float* Wq = (const float*)d_in[3];
    const float* bq = (const float*)d_in[4];
    const float* Wk = (const float*)d_in[5];
    const float* bk = (const float*)d_in[6];
    const float* Wv = (const float*)d_in[7];
    const float* bv = (const float*)d_in[8];
    const float* Wo = (const float*)d_in[9];
    const float* bo = (const float*)d_in[10];
    float* out = (float*)d_out;

    const size_t E = (size_t)ROWS * DMODEL;
    bf16* Qb  = (bf16*)d_ws;      // [row][col]      8 MB
    bf16* Kb  = Qb + E;           // [row][col]      8 MB
    bf16* VTb = Kb + E;           // [col][row] (V^T) 8 MB
    bf16* Ab  = VTb + E;          // [row][col]      8 MB

    const dim3 blk(256);
    const dim3 gGemm(ROWS / 128, DMODEL / 64);     // 32 x 16

    gemm_wmma<false, true, false><<<gGemm, blk, 0, stream>>>(queries, Wq, bq, Qb,
                                                             ROWS, DMODEL, DMODEL);
    gemm_wmma<false, true, false><<<gGemm, blk, 0, stream>>>(keys,    Wk, bk, Kb,
                                                             ROWS, DMODEL, DMODEL);
    // V projection stored transposed for the attention AV GEMM
    gemm_wmma<false, true, true ><<<gGemm, blk, 0, stream>>>(values,  Wv, bv, VTb,
                                                             ROWS, DMODEL, DMODEL);

    hopfield_attn<<<dim3(SEQ / 128, BATCH * NHEADS), blk, 0, stream>>>(Qb, Kb, VTb, Ab);

    gemm_wmma<true, false, false><<<gGemm, blk, 0, stream>>>(Ab, Wo, bo, out,
                                                             ROWS, DMODEL, DMODEL);
}
